// CRF_86337432584713
// MI455X (gfx1250) — compile-verified, matrix operand order checked by reading, not run on hardware
//
#include <hip/hip_runtime.h>
#include <hip/hip_bf16.h>
#include <cstdint>

typedef __attribute__((ext_vector_type(16))) _Float16 v16h;
typedef __attribute__((ext_vector_type(8)))  _Float16 v8h;
typedef __attribute__((ext_vector_type(8)))  float    v8f;
typedef __attribute__((ext_vector_type(4)))  float    v4f;

#define Tn 1024
#define Bn 128
#define Ln 128
#define ROWS 16      // batch rows per workgroup (WMMA M)
#define PMS 136      // Pm row stride (halves): 272B -> 4-bank stagger, conflict-free A loads
#define SCS 132      // sc row stride (floats): 528B -> 4-bank stagger
#define PBS 132      // pred staging row stride (floats), keeps 16B alignment (528B)

#define LOG2E 1.4426950408889634074f
#define LN2   0.69314718055994530942f

// Forward scan: per workgroup, 16 batch rows.  Each step:
//   m_b = max_j s[b,j];  P[b,i] = exp(s[b,i]-m_b) (f16)
//   Acc = P (16x128) * Et^T (128x16 per wave)        <- v_wmma_f32_16x16x32_f16
//   s'[b,j] = mask ? m_b + c_j + log(Acc) + pred[t,b,j] : s[b,j]   (branchless)
// pred tile for step t+1 is prefetched to LDS via global_load_async_to_lds_b128.
__global__ __launch_bounds__(256) void crf_forward_kernel(
    const float* __restrict__ pred,    // [T][B][L]
    const uint8_t* __restrict__ mask,  // [T][B]
    const float* __restrict__ trans,   // [L][L]
    const float* __restrict__ start,   // [L]
    const float* __restrict__ endsc,   // [L]
    float* __restrict__ denom)         // [B]
{
    __shared__ float    sc[ROWS][SCS];       // running scores (f32 mirror)
    __shared__ _Float16 Pm[ROWS][PMS];       // exp'd scores (A panel, f16)
    __shared__ _Float16 Et[Ln * Ln];         // Et[j][k] = exp(trans[k][j]-c[j])
    __shared__ float    cj[Ln];
    __shared__ __align__(16) float mrow[ROWS];
    __shared__ __align__(16) float predbuf[2][ROWS * PBS];  // async staging

    const int tid  = threadIdx.x;
    const int lane = tid & 31;
    const int wave = tid >> 5;               // 0..7 -> column tile
    const int b0   = blockIdx.x * ROWS;

    // ---- c[j] = max_k trans[k][j] ----
    if (tid < Ln) {
        float m = -3.0e38f;
        for (int k = 0; k < Ln; ++k) m = fmaxf(m, trans[k * Ln + tid]);
        cj[tid] = m;
    }
    __syncthreads();
    // ---- Et[j][k] = exp(trans[k][j] - c[j]) in f16 (values in (0,1]) ----
    for (int idx = tid; idx < Ln * Ln; idx += 256) {
        int k = idx >> 7, j = idx & (Ln - 1);
        Et[j * Ln + k] = (_Float16)__builtin_amdgcn_exp2f((trans[idx] - cj[j]) * LOG2E);
    }
    // ---- init scores: start[j] + pred[0][b][j] ----
    for (int idx = tid; idx < ROWS * Ln; idx += 256) {
        int r = idx >> 7, j = idx & (Ln - 1);
        sc[r][j] = start[j] + pred[(size_t)(b0 + r) * Ln + j];
    }
    __syncthreads();

    // ---- preload B fragments (time-invariant): wave w owns columns [16w,16w+16)
    const int nloc = lane & 15;
    const int khal = (lane >> 4) * 8;
    const int jcol = wave * 16 + nloc;
    v16h bfrag[4];
#pragma unroll
    for (int kk = 0; kk < 4; ++kk) {
        int kb = kk * 32;
        v8h lo = *(const v8h*)&Et[jcol * Ln + kb + khal];
        v8h hi = *(const v8h*)&Et[jcol * Ln + kb + 16 + khal];
#pragma unroll
        for (int e = 0; e < 8; ++e) { bfrag[kk][e] = lo[e]; bfrag[kk][e + 8] = hi[e]; }
    }

    const int arow   = lane & 15;            // A: row this lane holds
    const int vbaseM = (lane >> 4) * 8;      // C/D: M = vbaseM + v
    const float cterm = cj[jcol];            // time-invariant column term

    // running scores in C-fragment layout (row vbaseM+v, column jcol)
    float sreg[8];
#pragma unroll
    for (int v = 0; v < 8; ++v) sreg[v] = sc[vbaseM + v][jcol];

    const int rr = tid >> 4, ss = tid & 15;  // max/exp phase mapping

    // async-copy parameters: this thread moves 2 x b128 chunks per step
    const int cp_r0 = tid >> 5,         cp_j0 = (tid & 31) * 4;       // chunk 0
    const int cp_r1 = (tid + 256) >> 5, cp_j1 = (tid & 31) * 4;       // chunk 1
    const uint32_t pb0 = (uint32_t)(uintptr_t)&predbuf[0][0];
    const uint32_t pb1 = (uint32_t)(uintptr_t)&predbuf[1][0];

    // prologue: prefetch pred tile for t=1 into buffer 1
    {
        const float* gsrc = pred + ((size_t)1 * Bn + b0) * Ln;
        uint32_t l0 = pb1 + (uint32_t)((cp_r0 * PBS + cp_j0) * 4);
        uint32_t l1 = pb1 + (uint32_t)((cp_r1 * PBS + cp_j1) * 4);
        const float* g0 = gsrc + cp_r0 * Ln + cp_j0;
        const float* g1 = gsrc + cp_r1 * Ln + cp_j1;
        asm volatile("global_load_async_to_lds_b128 %0, %1, off" :: "v"(l0), "v"(g0) : "memory");
        asm volatile("global_load_async_to_lds_b128 %0, %1, off" :: "v"(l1), "v"(g1) : "memory");
    }
    uint2 mk = *(const uint2*)&mask[1 * Bn + b0 + vbaseM];   // mask bytes for t=1

    for (int t = 1; t < Tn; ++t) {
        // ---- prefetch step t+1 (clamped on last iteration, keeps counts uniform) ----
        const int tnext = (t + 1 < Tn) ? (t + 1) : (Tn - 1);
        {
            const uint32_t pbn = ((t + 1) & 1) ? pb1 : pb0;
            const float* gsrc = pred + ((size_t)tnext * Bn + b0) * Ln;
            uint32_t l0 = pbn + (uint32_t)((cp_r0 * PBS + cp_j0) * 4);
            uint32_t l1 = pbn + (uint32_t)((cp_r1 * PBS + cp_j1) * 4);
            const float* g0 = gsrc + cp_r0 * Ln + cp_j0;
            const float* g1 = gsrc + cp_r1 * Ln + cp_j1;
            asm volatile("global_load_async_to_lds_b128 %0, %1, off" :: "v"(l0), "v"(g0) : "memory");
            asm volatile("global_load_async_to_lds_b128 %0, %1, off" :: "v"(l1), "v"(g1) : "memory");
        }
        uint2 mk_next = *(const uint2*)&mask[(size_t)tnext * Bn + b0 + vbaseM];

        // ---- row max: 8-wide serial + 16-lane butterfly (no barrier) ----
        float m = -3.0e38f;
#pragma unroll
        for (int c = 0; c < 8; ++c) m = fmaxf(m, sc[rr][ss * 8 + c]);
#pragma unroll
        for (int d = 1; d < 16; d <<= 1) m = fmaxf(m, __shfl_xor(m, d, 32));
        mrow[rr] = m;                        // 16 lanes write same value
        // ---- exp'd A panel ----
#pragma unroll
        for (int c = 0; c < 8; ++c) {
            int j = ss * 8 + c;
            Pm[rr][j] = (_Float16)__builtin_amdgcn_exp2f((sc[rr][j] - m) * LOG2E);
        }
        // retire step t's async tile (the 2 just-issued for t+1 may stay in flight),
        // then barrier makes it visible to all waves
        asm volatile("s_wait_asynccnt 2" ::: "memory");
        __syncthreads();

        // ---- GEMM: this wave's 16x16 tile, K=128 in 4 WMMA steps ----
        v8f acc = {};
#pragma unroll
        for (int kk = 0; kk < 4; ++kk) {
            int kb = kk * 32;
            v8h lo = *(const v8h*)&Pm[arow][kb + khal];
            v8h hi = *(const v8h*)&Pm[arow][kb + 16 + khal];
            v16h a;
#pragma unroll
            for (int e = 0; e < 8; ++e) { a[e] = lo[e]; a[e + 8] = hi[e]; }
            acc = __builtin_amdgcn_wmma_f32_16x16x32_f16(
                false, a, false, bfrag[kk], (short)0, acc, false, false);
        }

        // ---- masked update: ALL operands computed unconditionally, then a pure
        //      register bit-blend (no load/log inside the select -> no branches) ----
        const float* pbuf = &predbuf[t & 1][0];
        float pv[8];
#pragma unroll
        for (int v = 0; v < 8; ++v) pv[v] = pbuf[(vbaseM + v) * PBS + jcol];
        v4f mv0 = *(const v4f*)&mrow[vbaseM];
        v4f mv1 = *(const v4f*)&mrow[vbaseM + 4];
        float news[8];
#pragma unroll
        for (int v = 0; v < 8; ++v) {
            float mrv = (v < 4) ? mv0[v] : mv1[v - 4];
            news[v] = mrv + cterm + __builtin_amdgcn_logf(acc[v]) * LN2 + pv[v];
        }
#pragma unroll
        for (int v = 0; v < 8; ++v) {
            uint32_t mbyte = (v < 4) ? ((mk.x >> (8 * v)) & 0xffu)
                                     : ((mk.y >> (8 * (v - 4))) & 0xffu);
            uint32_t sel = 0u - (uint32_t)(mbyte != 0u);     // all-ones if masked-in
            uint32_t nb  = __float_as_uint(news[v]);
            uint32_t ob  = __float_as_uint(sreg[v]);
            sreg[v] = __uint_as_float((nb & sel) | (ob & ~sel));
            sc[vbaseM + v][jcol] = sreg[v];
        }
        mk = mk_next;
        __syncthreads();
    }

    // denominator[b] = logsumexp_j(sc[b][j] + end[j])
    if (tid < ROWS) {
        int r = tid;
        float m = -3.0e38f;
        for (int j = 0; j < Ln; ++j) m = fmaxf(m, sc[r][j] + endsc[j]);
        float s = 0.f;
        for (int j = 0; j < Ln; ++j)
            s += __builtin_amdgcn_exp2f((sc[r][j] + endsc[j] - m) * LOG2E);
        denom[b0 + r] = m + __builtin_amdgcn_logf(s) * LN2;
    }
}

// numerator[b]: gold-path score (one thread per batch row)
__global__ __launch_bounds__(128) void crf_numerator_kernel(
    const float* __restrict__ pred, const int* __restrict__ targets,
    const uint8_t* __restrict__ mask, const float* __restrict__ trans,
    const float* __restrict__ start, const float* __restrict__ endsc,
    float* __restrict__ num)
{
    int b = threadIdx.x;
    int prev = targets[b];
    float acc = start[prev] + pred[(size_t)b * Ln + prev];
    int cnt = mask[b] ? 1 : 0;
    for (int t = 1; t < Tn; ++t) {
        int tg = targets[t * Bn + b];
        float mt = mask[t * Bn + b] ? 1.0f : 0.0f;
        acc += (trans[prev * Ln + tg] + pred[((size_t)t * Bn + b) * Ln + tg]) * mt;
        cnt += (mt != 0.0f);
        prev = tg;   // reference advances prev regardless of mask
    }
    int last = cnt - 1;
    acc += endsc[targets[last * Bn + b]];
    num[b] = acc;
}

__global__ __launch_bounds__(128) void crf_reduce_kernel(
    const float* __restrict__ denom, const float* __restrict__ num,
    float* __restrict__ out)
{
    __shared__ float red[Bn];
    int b = threadIdx.x;
    red[b] = denom[b] - num[b];
    __syncthreads();
    for (int s = Bn / 2; s > 0; s >>= 1) {
        if (b < s) red[b] += red[b + s];
        __syncthreads();
    }
    if (b == 0) out[0] = red[0] * (1.0f / Bn);
}

extern "C" void kernel_launch(void* const* d_in, const int* in_sizes, int n_in,
                              void* d_out, int out_size, void* d_ws, size_t ws_size,
                              hipStream_t stream) {
    const float*   pred    = (const float*)d_in[0];
    const int*     targets = (const int*)d_in[1];
    const uint8_t* mask    = (const uint8_t*)d_in[2];  // jnp bool -> 1 byte
    const float*   trans   = (const float*)d_in[3];
    const float*   start   = (const float*)d_in[4];
    const float*   endsc   = (const float*)d_in[5];

    float* denom = (float*)d_ws;       // [128]
    float* num   = denom + Bn;         // [128]

    crf_forward_kernel<<<Bn / ROWS, 256, 0, stream>>>(pred, mask, trans, start, endsc, denom);
    crf_numerator_kernel<<<1, Bn, 0, stream>>>(pred, targets, mask, trans, start, endsc, num);
    crf_reduce_kernel<<<1, Bn, 0, stream>>>(denom, num, (float*)d_out);
}
